// TransMIL_8426725834975
// MI455X (gfx1250) — compile-verified, matrix operand order checked by reading, not run on hardware
//
#include <hip/hip_runtime.h>
#include <hip/hip_bf16.h>

typedef __bf16 bf16;
typedef __attribute__((ext_vector_type(16))) __bf16 v16bf;
typedef __attribute__((ext_vector_type(8)))  float  v8f;
typedef __attribute__((ext_vector_type(4)))  int    v4i;

#define NPAD   16640   // 255 front-pad + cls + 16384 tokens
#define NTOK   16385   // cls + 16384
#define PADF   255
#define DIMC   512
#define QKVC   1536
#define NHEAD  8
#define DHD    64
#define LSUB   65      // NPAD / 256
#define FCHUNK 4160    // NPAD / 4 (flash split-K chunk)

#if __has_builtin(__builtin_amdgcn_global_load_async_to_lds_b128) && \
    __has_builtin(__builtin_amdgcn_s_wait_asynccnt)
#define HAVE_ASYNC_LDS 1
#define AS1 __attribute__((address_space(1)))
#define AS3 __attribute__((address_space(3)))
#endif

// ---------------------------------------------------------------------------
// helpers
// ---------------------------------------------------------------------------
__device__ inline v8f zero8() {
  v8f z;
#pragma unroll
  for (int i = 0; i < 8; ++i) z[i] = 0.f;
  return z;
}

// Load a 16x32 (A) or 32x16-as-N-major (B) bf16 WMMA fragment from an LDS row.
// rowptr points at the K-window start of this lane's row; cb = (lane/16)*8.
__device__ inline v16bf load_frag(const bf16* rowptr, int cb) {
  v16bf r;
#pragma unroll
  for (int j = 0; j < 8; ++j) {
    r[j]     = rowptr[cb + j];
    r[8 + j] = rowptr[cb + 16 + j];
  }
  return r;
}

union pack4 { bf16 h[4]; uint2 u; };

__device__ inline void pack_store4(bf16* dst, float4 v) {
  pack4 p;
  p.h[0] = (bf16)v.x; p.h[1] = (bf16)v.y;
  p.h[2] = (bf16)v.z; p.h[3] = (bf16)v.w;
  *(uint2*)dst = p.u;
}

// ---------------------------------------------------------------------------
// Generic batched GEMM:  D = alpha * A(MxK) @ B(KxN)  [+bias][relu][qscale]
// bf16 WMMA, f32 accumulate, double-buffered LDS, vectorized f32->bf16 staging.
// REQUIRES: M % 128 == 0, K % 32 == 0, N % 4 == 0 (all call sites satisfy).
// flags: bit0 = relu, bit1 = scale cols < 512 by 0.125 (q scaling in QKV gemm)
// ---------------------------------------------------------------------------
__device__ inline void stage_tile(bf16 (*As)[32], bf16 (*Bs)[32],
                                  const float* __restrict__ A,
                                  const float* __restrict__ B,
                                  int m0, int n0, int k0,
                                  int lda, int ldb, int N, int tid)
{
  float4 va[4], vb[4];
  // phase 1: issue all A global loads (keeps loads in flight)
#pragma unroll
  for (int q = 0; q < 4; ++q) {          // A tile 128x32, 1024 float4
    int vecid = q * 256 + tid;
    int r = vecid >> 3, c4 = (vecid & 7) * 4;
    va[q] = *(const float4*)&A[(long)(m0 + r) * lda + k0 + c4];
  }
  if (n0 + 128 <= N) {
    // uniform fast path: full B tile, no guards
#pragma unroll
    for (int q = 0; q < 4; ++q) {        // B tile 32x128
      int vecid = q * 256 + tid;
      int kk = vecid >> 5, nn4 = (vecid & 31) * 4;
      vb[q] = *(const float4*)&B[(long)(k0 + kk) * ldb + n0 + nn4];
    }
#pragma unroll
    for (int q = 0; q < 4; ++q) {
      int vecid = q * 256 + tid;
      int r = vecid >> 3, c4 = (vecid & 7) * 4;
      pack_store4(&As[r][c4], va[q]);
    }
#pragma unroll
    for (int q = 0; q < 4; ++q) {        // B -> N-major scatter
      int vecid = q * 256 + tid;
      int kk = vecid >> 5, nn4 = (vecid & 31) * 4;
      Bs[nn4 + 0][kk] = (bf16)vb[q].x;
      Bs[nn4 + 1][kk] = (bf16)vb[q].y;
      Bs[nn4 + 2][kk] = (bf16)vb[q].z;
      Bs[nn4 + 3][kk] = (bf16)vb[q].w;
    }
  } else {
    // edge path (only the N=64 z@bo GEMM): per-vec4 guards
    bool bg[4];
#pragma unroll
    for (int q = 0; q < 4; ++q) {
      int vecid = q * 256 + tid;
      int kk = vecid >> 5, nn4 = (vecid & 31) * 4;
      bg[q] = (n0 + nn4 < N);
      if (bg[q]) vb[q] = *(const float4*)&B[(long)(k0 + kk) * ldb + n0 + nn4];
    }
#pragma unroll
    for (int q = 0; q < 4; ++q) {
      int vecid = q * 256 + tid;
      int r = vecid >> 3, c4 = (vecid & 7) * 4;
      pack_store4(&As[r][c4], va[q]);
    }
#pragma unroll
    for (int q = 0; q < 4; ++q) {
      int vecid = q * 256 + tid;
      int kk = vecid >> 5, nn4 = (vecid & 31) * 4;
      if (bg[q]) {
        Bs[nn4 + 0][kk] = (bf16)vb[q].x;
        Bs[nn4 + 1][kk] = (bf16)vb[q].y;
        Bs[nn4 + 2][kk] = (bf16)vb[q].z;
        Bs[nn4 + 3][kk] = (bf16)vb[q].w;
      }
    }
  }
}

__global__ __launch_bounds__(256) void gemm_bf16_kernel(
    const float* __restrict__ A, const float* __restrict__ B,
    float* __restrict__ D, int M, int N, int K, int lda, int ldb, int ldc,
    long sA, long sB, long sD, const float* __restrict__ bias, float alpha,
    int flags)
{
  const int bz = blockIdx.z;
  A += (long)bz * sA; B += (long)bz * sB; D += (long)bz * sD;

  __shared__ bf16 As[2][128][32];   // 16 KB
  __shared__ bf16 Bs[2][128][32];   // 16 KB (N-major)

  const int tid  = threadIdx.x;
  const int lane = tid & 31;
  const int wid  = tid >> 5;
  const int wm   = wid & 3;      // 4 wave-rows of 32
  const int wn   = wid >> 2;     // 2 wave-cols of 64
  const int m0   = blockIdx.y * 128;
  const int n0   = blockIdx.x * 128;
  const int half = lane >> 4;
  const int l16  = lane & 15;
  const int cb   = half * 8;

  v8f acc[2][4];
#pragma unroll
  for (int i = 0; i < 2; ++i)
#pragma unroll
    for (int j = 0; j < 4; ++j) acc[i][j] = zero8();

  const int nk = K >> 5;
  stage_tile(As[0], Bs[0], A, B, m0, n0, 0, lda, ldb, N, tid);

  for (int kt = 0; kt < nk; ++kt) {
    __syncthreads();
    if (kt + 1 < nk)
      stage_tile(As[(kt + 1) & 1], Bs[(kt + 1) & 1], A, B, m0, n0,
                 (kt + 1) * 32, lda, ldb, N, tid);
    const int cur = kt & 1;
    v16bf af[2], bfr[4];
#pragma unroll
    for (int i = 0; i < 2; ++i)
      af[i] = load_frag(&As[cur][wm * 32 + i * 16 + l16][0], cb);
#pragma unroll
    for (int j = 0; j < 4; ++j)
      bfr[j] = load_frag(&Bs[cur][wn * 64 + j * 16 + l16][0], cb);
#pragma unroll
    for (int i = 0; i < 2; ++i)
#pragma unroll
      for (int j = 0; j < 4; ++j)
        acc[i][j] = __builtin_amdgcn_wmma_f32_16x16x32_bf16(
            false, af[i], false, bfr[j], (short)0, acc[i][j], false, false);
  }

  // epilogue (C layout: VGPR vi, lanes 0-15 -> M=vi, lanes 16-31 -> M=vi+8)
#pragma unroll
  for (int i = 0; i < 2; ++i)
#pragma unroll
    for (int j = 0; j < 4; ++j) {
      int col = n0 + wn * 64 + j * 16 + l16;
      if (col < N) {
        float badd = bias ? bias[col] : 0.f;
        float csc = ((flags & 2) && col < 512) ? 0.125f : 1.f;
#pragma unroll
        for (int vi = 0; vi < 8; ++vi) {
          int row = m0 + wm * 32 + i * 16 + vi + 8 * half;
          float v = acc[i][j][vi] * alpha * csc + badd;
          if (flags & 1) v = fmaxf(v, 0.f);
          D[(long)row * ldc + col] = v;
        }
      }
    }
}

// ---------------------------------------------------------------------------
// Fused a1 @ M:  per (head, 64-row tile): S = q*kl^T (WMMA), row softmax,
// P @ M (WMMA) -> attn_cat[n][head*64+d].  4 waves, 128 threads.
// ---------------------------------------------------------------------------
__global__ __launch_bounds__(128) void attn_pipeline_kernel(
    const float* __restrict__ qkv, const float* __restrict__ kl,
    const float* __restrict__ Mm, float* __restrict__ attn_cat)
{
  const int head = blockIdx.y;
  const int rt   = blockIdx.x;            // 0..259
  __shared__ bf16 shKM[256 * 64];         // kl[256][64], later Ms[64][256]
  __shared__ bf16 shP[4][16 * 256];       // q (phase1, first 1024) then P

  const int tid  = threadIdx.x;
  const int lane = tid & 31;
  const int wid  = tid >> 5;              // 0..3, 16 rows each
  const int half = lane >> 4;
  const int l16  = lane & 15;
  const int cb   = half * 8;

  // stage kl (256x64) as bf16 (4096 float4, 128 threads -> 32 each)
#pragma unroll 8
  for (int t = 0; t < 32; ++t) {
    int vecid = t * 128 + tid;
    float4 v = *(const float4*)&kl[(long)head * 16384 + vecid * 4];
    pack_store4(&shKM[vecid * 4], v);
  }
  // stage this wave's 16x64 q rows (scaled q already in qkv cols [0,512))
  bf16* qs = &shP[wid][0];
#pragma unroll
  for (int t = 0; t < 8; ++t) {
    int vecid = t * 32 + lane;            // 256 float4
    int r = vecid >> 4, c4 = (vecid & 15) * 4;
    float4 v = *(const float4*)
        &qkv[(long)(rt * 64 + wid * 16 + r) * QKVC + head * 64 + c4];
    pack_store4(&qs[r * 64 + c4], v);
  }
  __syncthreads();

  // S = q @ kl^T : 16 rows x 256 cols in 16 accumulator tiles
  v8f S[16];
#pragma unroll
  for (int nt = 0; nt < 16; ++nt) S[nt] = zero8();
#pragma unroll
  for (int ks = 0; ks < 2; ++ks) {
    v16bf a = load_frag(&qs[l16 * 64 + ks * 32], cb);
#pragma unroll
    for (int nt = 0; nt < 16; ++nt) {
      v16bf b = load_frag(&shKM[(nt * 16 + l16) * 64 + ks * 32], cb);
      S[nt] = __builtin_amdgcn_wmma_f32_16x16x32_bf16(
          false, a, false, b, (short)0, S[nt], false, false);
    }
  }

  // row softmax over 256 cols (lanes within each 16-lane half hold one row)
  float rsum[8];
#pragma unroll
  for (int vi = 0; vi < 8; ++vi) {
    float m = -3.0e38f;
#pragma unroll
    for (int nt = 0; nt < 16; ++nt) m = fmaxf(m, S[nt][vi]);
    m = fmaxf(m, __shfl_xor(m, 1));
    m = fmaxf(m, __shfl_xor(m, 2));
    m = fmaxf(m, __shfl_xor(m, 4));
    m = fmaxf(m, __shfl_xor(m, 8));
    float s = 0.f;
#pragma unroll
    for (int nt = 0; nt < 16; ++nt) {
      float e = __expf(S[nt][vi] - m);
      S[nt][vi] = e;
      s += e;
    }
    s += __shfl_xor(s, 1);
    s += __shfl_xor(s, 2);
    s += __shfl_xor(s, 4);
    s += __shfl_xor(s, 8);
    rsum[vi] = s;
  }
  // write P (bf16) into this wave's LDS region (q no longer needed)
#pragma unroll
  for (int nt = 0; nt < 16; ++nt)
#pragma unroll
    for (int vi = 0; vi < 8; ++vi)
      shP[wid][(vi + 8 * half) * 256 + nt * 16 + l16] =
          (bf16)(S[nt][vi] / rsum[vi]);
  __syncthreads();

  // stage Ms[n][k] = M[head][k][n]  (64 x 256, N-major; 4096 float4)
#pragma unroll 8
  for (int t = 0; t < 32; ++t) {
    int vecid = t * 128 + tid;
    int kk = vecid >> 4, nn4 = (vecid & 15) * 4;
    float4 v = *(const float4*)&Mm[((long)head * 256 + kk) * 64 + nn4];
    shKM[(nn4 + 0) * 256 + kk] = (bf16)v.x;
    shKM[(nn4 + 1) * 256 + kk] = (bf16)v.y;
    shKM[(nn4 + 2) * 256 + kk] = (bf16)v.z;
    shKM[(nn4 + 3) * 256 + kk] = (bf16)v.w;
  }
  __syncthreads();

  // O = P(16x256) @ M(256x64)
  v8f O[4];
#pragma unroll
  for (int j = 0; j < 4; ++j) O[j] = zero8();
#pragma unroll
  for (int ks = 0; ks < 8; ++ks) {
    v16bf a = load_frag(&shP[wid][l16 * 256 + ks * 32], cb);
#pragma unroll
    for (int j = 0; j < 4; ++j) {
      v16bf b = load_frag(&shKM[(j * 16 + l16) * 256 + ks * 32], cb);
      O[j] = __builtin_amdgcn_wmma_f32_16x16x32_bf16(
          false, a, false, b, (short)0, O[j], false, false);
    }
  }
#pragma unroll
  for (int j = 0; j < 4; ++j)
#pragma unroll
    for (int vi = 0; vi < 8; ++vi) {
      int row = rt * 64 + wid * 16 + vi + 8 * half;
      int col = head * 64 + j * 16 + l16;
      attn_cat[(long)row * DIMC + col] = O[j][vi];
    }
}

// ---------------------------------------------------------------------------
// flash-style a3 @ v, split over 4 key chunks per head for occupancy.
// k/v tiles staged via gfx1250 async global->LDS DMA when available.
// partial: unnormalized acc + running (max, sum) per (head, chunk, query).
// ---------------------------------------------------------------------------
__global__ __launch_bounds__(256) void flash_a3v_partial_kernel(
    const float* __restrict__ qkv, const float* __restrict__ ql,
    float* __restrict__ pacc, float* __restrict__ pm, float* __restrict__ ps)
{
  const int head  = blockIdx.x;
  const int chunk = blockIdx.y;
  const int i     = threadIdx.x;
  float q[DHD];
#pragma unroll
  for (int d = 0; d < DHD; ++d) q[d] = ql[((long)head * 256 + i) * DHD + d];

  __shared__ float ks[64][64];
  __shared__ float vs[64][64];
  float acc[DHD];
#pragma unroll
  for (int d = 0; d < DHD; ++d) acc[d] = 0.f;
  float m = -3.0e38f, ssum = 0.f;

  const int nb0 = chunk * FCHUNK;
  for (int nb = nb0; nb < nb0 + FCHUNK; nb += 64) {
#ifdef HAVE_ASYNC_LDS
#pragma unroll
    for (int t = 0; t < 4; ++t) {        // 1024 float4 per matrix
      int vecid = t * 256 + i;
      int r = vecid >> 4, c4 = (vecid & 15) * 4;
      __builtin_amdgcn_global_load_async_to_lds_b128(
          (AS1 v4i*)&qkv[(long)(nb + r) * QKVC + 512 + head * 64 + c4],
          (AS3 v4i*)&ks[r][c4], 0, 0);
      __builtin_amdgcn_global_load_async_to_lds_b128(
          (AS1 v4i*)&qkv[(long)(nb + r) * QKVC + 1024 + head * 64 + c4],
          (AS3 v4i*)&vs[r][c4], 0, 0);
    }
    __builtin_amdgcn_s_wait_asynccnt(0);
    __syncthreads();
#else
#pragma unroll
    for (int t = 0; t < 4; ++t) {        // 1024 float4 per matrix
      int vecid = t * 256 + i;
      int r = vecid >> 4, c4 = (vecid & 15) * 4;
      *(float4*)&ks[r][c4] =
          *(const float4*)&qkv[(long)(nb + r) * QKVC + 512 + head * 64 + c4];
      *(float4*)&vs[r][c4] =
          *(const float4*)&qkv[(long)(nb + r) * QKVC + 1024 + head * 64 + c4];
    }
    __syncthreads();
#endif
#pragma unroll 4
    for (int r = 0; r < 64; ++r) {
      float s = 0.f;
#pragma unroll
      for (int d = 0; d < DHD; ++d) s += q[d] * ks[r][d];
      float mn = fmaxf(m, s);
      float corr = __expf(m - mn);
      float p = __expf(s - mn);
      ssum = ssum * corr + p;
#pragma unroll
      for (int d = 0; d < DHD; ++d) acc[d] = acc[d] * corr + p * vs[r][d];
      m = mn;
    }
    __syncthreads();
  }
  long pidx = (long)(head * 4 + chunk) * 256 + i;
  pm[pidx] = m;
  ps[pidx] = ssum;
#pragma unroll
  for (int d = 0; d < DHD; ++d) pacc[pidx * DHD + d] = acc[d];
}

__global__ void flash_a3v_merge_kernel(
    const float* __restrict__ pacc, const float* __restrict__ pm,
    const float* __restrict__ ps, float* __restrict__ bo)
{
  int h = blockIdx.x >> 8, i = blockIdx.x & 255, d = threadIdx.x;
  float M = -3.0e38f;
#pragma unroll
  for (int c = 0; c < 4; ++c)
    M = fmaxf(M, pm[(long)(h * 4 + c) * 256 + i]);
  float tot = 0.f, val = 0.f;
#pragma unroll
  for (int c = 0; c < 4; ++c) {
    long pidx = (long)(h * 4 + c) * 256 + i;
    float w = __expf(pm[pidx] - M);
    tot += ps[pidx] * w;
    val += pacc[pidx * DHD + d] * w;
  }
  bo[((long)(h << 8) + i) * DHD + d] = val / tot;
}

// ---------------------------------------------------------------------------
// small kernels
// ---------------------------------------------------------------------------
__global__ void landmarks_kernel(const float* __restrict__ qkv,
                                 float* __restrict__ ql, float* __restrict__ kl)
{
  int h = blockIdx.x >> 8, i = blockIdx.x & 255, d = threadIdx.x;
  float sq = 0.f, sk = 0.f;
  for (int j = 0; j < LSUB; ++j) {
    long row = (long)i * LSUB + j;
    sq += qkv[row * QKVC + h * 64 + d];
    sk += qkv[row * QKVC + 512 + h * 64 + d];
  }
  ql[((long)(h << 8) + i) * 64 + d] = sq * (1.f / (float)LSUB);
  kl[((long)(h << 8) + i) * 64 + d] = sk * (1.f / (float)LSUB);
}

__global__ __launch_bounds__(256) void a2_softmax_kernel(
    const float* __restrict__ ql, const float* __restrict__ kl,
    float* __restrict__ a2)
{
  int h = blockIdx.x >> 8, i = blockIdx.x & 255, j = threadIdx.x;
  float s = 0.f;
#pragma unroll
  for (int d = 0; d < 64; ++d)
    s += ql[((long)(h << 8) + i) * 64 + d] * kl[((long)(h << 8) + j) * 64 + d];
  __shared__ float red[256];
  red[j] = s; __syncthreads();
  for (int off = 128; off; off >>= 1) {
    if (j < off) red[j] = fmaxf(red[j], red[j + off]);
    __syncthreads();
  }
  float mx = red[0]; __syncthreads();
  float e = __expf(s - mx);
  red[j] = e; __syncthreads();
  for (int off = 128; off; off >>= 1) {
    if (j < off) red[j] += red[j + off];
    __syncthreads();
  }
  a2[(((long)(h << 8) + i) << 8) + j] = e / red[0];
}

__global__ void scal_init_kernel(unsigned* s) { s[0] = 0u; s[1] = 0u; }

__global__ __launch_bounds__(256) void abs_rowsum_max_kernel(
    const float* __restrict__ a2, unsigned* out)
{
  int h = blockIdx.x >> 8, i = blockIdx.x & 255, j = threadIdx.x;
  float v = fabsf(a2[(((long)(h << 8) + i) << 8) + j]);
  __shared__ float red[256];
  red[j] = v; __syncthreads();
  for (int off = 128; off; off >>= 1) {
    if (j < off) red[j] += red[j + off];
    __syncthreads();
  }
  if (j == 0) atomicMax(out, __float_as_uint(red[0]));
}

__global__ __launch_bounds__(256) void abs_colsum_max_kernel(
    const float* __restrict__ a2, unsigned* out)
{
  int h = blockIdx.x >> 8, j = blockIdx.x & 255, i = threadIdx.x;
  float v = fabsf(a2[(((long)(h << 8) + i) << 8) + j]);
  __shared__ float red[256];
  red[i] = v; __syncthreads();
  for (int off = 128; off; off >>= 1) {
    if (i < off) red[i] += red[i + off];
    __syncthreads();
  }
  if (i == 0) atomicMax(out, __float_as_uint(red[0]));
}

__global__ __launch_bounds__(256) void pinv_z0_kernel(
    const float* __restrict__ a2, const unsigned* __restrict__ s,
    float* __restrict__ z)
{
  long idx = (long)blockIdx.x * 256 + threadIdx.x;
  int j = idx & 255, i = (idx >> 8) & 255, h = (int)(idx >> 16);
  float denom = __uint_as_float(s[0]) * __uint_as_float(s[1]);
  z[idx] = a2[((long)h << 16) | ((long)j << 8) | (long)i] / denom;
}

__global__ __launch_bounds__(256) void c_i_minus_kernel(
    const float* __restrict__ src, float* __restrict__ dst, float c)
{
  long idx = (long)blockIdx.x * 256 + threadIdx.x;
  int j = idx & 255, i = (idx >> 8) & 255;
  dst[idx] = (i == j ? c : 0.f) - src[idx];
}

__global__ __launch_bounds__(256) void conv_res_kernel(
    const float* __restrict__ qkv, const float* __restrict__ rw,
    float* __restrict__ attn_cat)
{
  long idx = (long)blockIdx.x * 256 + threadIdx.x;   // < 16640*512
  int c = (int)(idx & 511);
  long n = idx >> 9;
  int h = c >> 6;
  float s = 0.f;
#pragma unroll
  for (int j = 0; j < 33; ++j) {
    long r = n + j - 16;
    if (r >= 0 && r < NPAD) s += qkv[r * QKVC + 1024 + c] * rw[h * 33 + j];
  }
  attn_cat[idx] += s;
}

__global__ __launch_bounds__(256) void layernorm_rows_kernel(
    const float* __restrict__ x, float* __restrict__ y,
    const float* __restrict__ g, const float* __restrict__ b,
    int src_row0, int dst_row0)
{
  const float* xr = x + (long)(src_row0 + blockIdx.x) * DIMC;
  float* yr = y + (long)(dst_row0 + blockIdx.x) * DIMC;
  int t = threadIdx.x;
  float v0 = xr[t], v1 = xr[t + 256];
  __shared__ float red[256];
  red[t] = v0 + v1; __syncthreads();
  for (int off = 128; off; off >>= 1) {
    if (t < off) red[t] += red[t + off];
    __syncthreads();
  }
  float mu = red[0] * (1.f / 512.f); __syncthreads();
  float d0 = v0 - mu, d1 = v1 - mu;
  red[t] = d0 * d0 + d1 * d1; __syncthreads();
  for (int off = 128; off; off >>= 1) {
    if (t < off) red[t] += red[t + off];
    __syncthreads();
  }
  float rs = rsqrtf(red[0] * (1.f / 512.f) + 1e-5f);
  yr[t]       = d0 * rs * g[t] + b[t];
  yr[t + 256] = d1 * rs * g[t + 256] + b[t + 256];
}

__global__ void zero_rows_kernel(float* __restrict__ y) {
  long base = (long)blockIdx.x * DIMC;
  y[base + threadIdx.x] = 0.f;
  y[base + threadIdx.x + 256] = 0.f;
}

__global__ void cls_copy_kernel(float* __restrict__ h,
                                const float* __restrict__ cls) {
  h[(long)PADF * DIMC + threadIdx.x] = cls[threadIdx.x];
}

__global__ __launch_bounds__(256) void vec_add_kernel(
    float* __restrict__ dst, const float* __restrict__ src) {
  long idx = (long)blockIdx.x * 256 + threadIdx.x;
  dst[idx] += src[idx];
}

__global__ __launch_bounds__(256) void copy_to_h_kernel(
    float* __restrict__ h, const float* __restrict__ src) {
  long idx = (long)blockIdx.x * 256 + threadIdx.x;   // < 16384*512
  h[(long)(PADF + 1) * DIMC + idx] = src[idx];
}

__global__ __launch_bounds__(128) void ppeg_conv_kernel(
    const float* __restrict__ h, const float* __restrict__ w7,
    const float* __restrict__ b7, const float* __restrict__ w5,
    const float* __restrict__ b5, const float* __restrict__ w3,
    const float* __restrict__ b3, float* __restrict__ out)
{
  int pix = blockIdx.x;              // 0..16383
  int y = pix >> 7, x = pix & 127;
  int c = blockIdx.y * 128 + threadIdx.x;
  const long tok0 = (long)(PADF + 1) * DIMC;
  float acc = h[tok0 + (long)pix * DIMC + c] + b7[c] + b5[c] + b3[c];
#pragma unroll
  for (int ky = -3; ky <= 3; ++ky) {
#pragma unroll
    for (int kx = -3; kx <= 3; ++kx) {
      int yy = y + ky, xx = x + kx;
      if (yy >= 0 && yy < 128 && xx >= 0 && xx < 128) {
        float v = h[tok0 + (long)(yy * 128 + xx) * DIMC + c];
        acc += v * w7[c * 49 + (ky + 3) * 7 + (kx + 3)];
        if (ky >= -2 && ky <= 2 && kx >= -2 && kx <= 2)
          acc += v * w5[c * 25 + (ky + 2) * 5 + (kx + 2)];
        if (ky >= -1 && ky <= 1 && kx >= -1 && kx <= 1)
          acc += v * w3[c * 9 + (ky + 1) * 3 + (kx + 1)];
      }
    }
  }
  out[(long)pix * DIMC + c] = acc;
}

__global__ void final_head_kernel(const float* __restrict__ ln,
                                  const float* __restrict__ w,
                                  const float* __restrict__ b,
                                  float* __restrict__ out)
{
  int o = threadIdx.x >> 5, lane = threadIdx.x & 31;
  float s = 0.f;
  for (int k = lane; k < 512; k += 32) s += ln[k] * w[k * 4 + o];
  s += __shfl_xor(s, 1);
  s += __shfl_xor(s, 2);
  s += __shfl_xor(s, 4);
  s += __shfl_xor(s, 8);
  s += __shfl_xor(s, 16);
  if (lane == 0) out[o] = s + b[o];
}

// ---------------------------------------------------------------------------
// host orchestration
// ---------------------------------------------------------------------------
static void gemm(hipStream_t s, const float* A, const float* B, float* D,
                 int M, int N, int K, int lda, int ldb, int ldc,
                 long sA, long sB, long sD, int batch,
                 const float* bias, float alpha, int flags)
{
  dim3 g((N + 127) / 128, M / 128, batch);
  gemm_bf16_kernel<<<g, 256, 0, s>>>(A, B, D, M, N, K, lda, ldb, ldc,
                                     sA, sB, sD, bias, alpha, flags);
}

static void attention_stage(hipStream_t s, float* h, float* x_ln, float* qkv,
    float* attn_cat, float* tmp, float* ql, float* kl, float* a2,
    float* xz, float* ta, float* tb, float* zA, float* zB, float* bo,
    float* Mm, float* pacc, float* pm, float* ps, unsigned* scal,
    const float* ng, const float* nb, const float* qkv_w,
    const float* out_w, const float* out_b, const float* res_w)
{
  zero_rows_kernel<<<PADF, 256, 0, s>>>(x_ln);
  layernorm_rows_kernel<<<NTOK, 256, 0, s>>>(h, x_ln, ng, nb, PADF, PADF);
  gemm(s, x_ln, qkv_w, qkv, NPAD, QKVC, DIMC, DIMC, QKVC, QKVC,
       0, 0, 0, 1, nullptr, 1.f, 2 /*q-scale*/);
  landmarks_kernel<<<2048, 64, 0, s>>>(qkv, ql, kl);
  a2_softmax_kernel<<<2048, 256, 0, s>>>(ql, kl, a2);
  scal_init_kernel<<<1, 1, 0, s>>>(scal);
  abs_rowsum_max_kernel<<<2048, 256, 0, s>>>(a2, scal + 0);
  abs_colsum_max_kernel<<<2048, 256, 0, s>>>(a2, scal + 1);
  pinv_z0_kernel<<<2048, 256, 0, s>>>(a2, scal, zA);
  for (int it = 0; it < 6; ++it) {
    float* zi = (it & 1) ? zB : zA;
    float* zo = (it & 1) ? zA : zB;
    gemm(s, a2, zi, xz, 256, 256, 256, 256, 256, 256,
         65536, 65536, 65536, NHEAD, nullptr, 1.f, 0);
    c_i_minus_kernel<<<2048, 256, 0, s>>>(xz, ta, 7.f);
    gemm(s, xz, ta, tb, 256, 256, 256, 256, 256, 256,
         65536, 65536, 65536, NHEAD, nullptr, 1.f, 0);
    c_i_minus_kernel<<<2048, 256, 0, s>>>(tb, ta, 15.f);
    gemm(s, xz, ta, tb, 256, 256, 256, 256, 256, 256,
         65536, 65536, 65536, NHEAD, nullptr, 1.f, 0);
    c_i_minus_kernel<<<2048, 256, 0, s>>>(tb, ta, 13.f);
    gemm(s, zi, ta, zo, 256, 256, 256, 256, 256, 256,
         65536, 65536, 65536, NHEAD, nullptr, 0.25f, 0);
  }
  dim3 gf(NHEAD, 4);
  flash_a3v_partial_kernel<<<gf, 256, 0, s>>>(qkv, ql, pacc, pm, ps);
  flash_a3v_merge_kernel<<<2048, 64, 0, s>>>(pacc, pm, ps, bo);
  gemm(s, zA, bo, Mm, 256, 64, 256, 256, 64, 64,
       65536, 16384, 16384, NHEAD, nullptr, 1.f, 0);
  dim3 ga(NPAD / 64, NHEAD);
  attn_pipeline_kernel<<<ga, 128, 0, s>>>(qkv, kl, Mm, attn_cat);
  conv_res_kernel<<<(NPAD * DIMC) / 256, 256, 0, s>>>(qkv, res_w, attn_cat);
  gemm(s, attn_cat, out_w, tmp, NPAD, DIMC, DIMC, DIMC, DIMC, DIMC,
       0, 0, 0, 1, out_b, 1.f, 0);
  vec_add_kernel<<<(NPAD * DIMC) / 256, 256, 0, s>>>(h, tmp);
}

extern "C" void kernel_launch(void* const* d_in, const int* in_sizes, int n_in,
                              void* d_out, int out_size, void* d_ws,
                              size_t ws_size, hipStream_t stream)
{
  const float* data_x    = (const float*)d_in[0];
  const float* fc1_w     = (const float*)d_in[1];
  const float* fc1_b     = (const float*)d_in[2];
  const float* cls_token = (const float*)d_in[3];
  const float* l1_ng     = (const float*)d_in[4];
  const float* l1_nb     = (const float*)d_in[5];
  const float* l1_qkv_w  = (const float*)d_in[6];
  const float* l1_out_w  = (const float*)d_in[7];
  const float* l1_out_b  = (const float*)d_in[8];
  const float* l1_res_w  = (const float*)d_in[9];
  const float* ppeg_w7   = (const float*)d_in[10];
  const float* ppeg_b7   = (const float*)d_in[11];
  const float* ppeg_w5   = (const float*)d_in[12];
  const float* ppeg_b5   = (const float*)d_in[13];
  const float* ppeg_w3   = (const float*)d_in[14];
  const float* ppeg_b3   = (const float*)d_in[15];
  const float* l2_ng     = (const float*)d_in[16];
  const float* l2_nb     = (const float*)d_in[17];
  const float* l2_qkv_w  = (const float*)d_in[18];
  const float* l2_out_w  = (const float*)d_in[19];
  const float* l2_out_b  = (const float*)d_in[20];
  const float* l2_res_w  = (const float*)d_in[21];
  const float* norm_g    = (const float*)d_in[22];
  const float* norm_b    = (const float*)d_in[23];
  const float* fc2_w     = (const float*)d_in[24];
  const float* fc2_b     = (const float*)d_in[25];

  float* ws = (float*)d_ws;
  size_t off = 0;
  auto alloc = [&](size_t n) { float* p = ws + off; off += n; return p; };
  float* h        = alloc((size_t)NPAD * DIMC);
  float* x_ln     = alloc((size_t)NPAD * DIMC);
  float* qkv      = alloc((size_t)NPAD * QKVC);
  float* attn_cat = alloc((size_t)NPAD * DIMC);
  float* tmp      = alloc((size_t)NPAD * DIMC);
  float* ql       = alloc((size_t)NHEAD * 256 * 64);
  float* kl       = alloc((size_t)NHEAD * 256 * 64);
  float* a2       = alloc((size_t)NHEAD * 256 * 256);
  float* xz       = alloc((size_t)NHEAD * 256 * 256);
  float* ta       = alloc((size_t)NHEAD * 256 * 256);
  float* tb       = alloc((size_t)NHEAD * 256 * 256);
  float* zA       = alloc((size_t)NHEAD * 256 * 256);
  float* zB       = alloc((size_t)NHEAD * 256 * 256);
  float* bo       = alloc((size_t)NHEAD * 256 * 64);
  float* Mm       = alloc((size_t)NHEAD * 256 * 64);
  float* pacc     = alloc((size_t)NHEAD * 4 * 256 * 64);
  float* pm       = alloc((size_t)NHEAD * 4 * 256);
  float* ps       = alloc((size_t)NHEAD * 4 * 256);
  float* ln_cls   = alloc(512);
  unsigned* scal  = (unsigned*)alloc(16);

  // fc1 + relu -> token rows of h (rows 256..16639)
  gemm(stream, data_x, fc1_w, h + (size_t)(PADF + 1) * DIMC,
       16384, DIMC, 1024, 1024, DIMC, DIMC, 0, 0, 0, 1, fc1_b, 1.f, 1);
  cls_copy_kernel<<<1, 512, 0, stream>>>(h, cls_token);

  attention_stage(stream, h, x_ln, qkv, attn_cat, tmp, ql, kl, a2, xz, ta, tb,
                  zA, zB, bo, Mm, pacc, pm, ps, scal, l1_ng, l1_nb, l1_qkv_w,
                  l1_out_w, l1_out_b, l1_res_w);

  dim3 gp(16384, 4);
  ppeg_conv_kernel<<<gp, 128, 0, stream>>>(h, ppeg_w7, ppeg_b7, ppeg_w5,
                                           ppeg_b5, ppeg_w3, ppeg_b3,
                                           attn_cat);
  copy_to_h_kernel<<<(16384 * DIMC) / 256, 256, 0, stream>>>(h, attn_cat);

  attention_stage(stream, h, x_ln, qkv, attn_cat, tmp, ql, kl, a2, xz, ta, tb,
                  zA, zB, bo, Mm, pacc, pm, ps, scal, l2_ng, l2_nb, l2_qkv_w,
                  l2_out_w, l2_out_b, l2_res_w);

  layernorm_rows_kernel<<<1, 256, 0, stream>>>(h, ln_cls, norm_g, norm_b,
                                               PADF, 0);
  final_head_kernel<<<1, 128, 0, stream>>>(ln_cls, fc2_w, fc2_b,
                                           (float*)d_out);
}
